// RFA_SPDConv_Fourier_17678085390722
// MI455X (gfx1250) — matrix operand order repaired and checked by reading, not compiled, not run on hardware
//
#include <hip/hip_runtime.h>
#include <hip/hip_bf16.h>
#include <math.h>

typedef __attribute__((ext_vector_type(16))) _Float16 v16h;
typedef __attribute__((ext_vector_type(8)))  float    v8f;

union FragH { v16h v; unsigned u[8]; };

__device__ __forceinline__ float fast_sigmoid(float v) {
  return __builtin_amdgcn_rcpf(1.f + __expf(-v));
}

// ============================================================================
// Generic tiled WMMA GEMM:  C[M x N] = epilogue( A[M x K] * B[K x N] )
//   - 256 threads = 8 wave32s; block tile BM=128, BN=64, BK=32
//   - double-buffered LDS software pipeline (1 barrier / K-step)
//   - waves arranged 4(M) x 2(N); each wave owns 32x32 via 2x2 fragments
//   - BMODE: 0 = B strided row-major (K x N), 1 = implicit-GEMM gather for
//            3x3 stride-2 pad-1 conv over xs_h (256 x 128 x 128) per batch
//   - CMODE: 0 = f32 store, 1 = f16 store, 2 = f16 Fourier scatter into `cat`
//   - ACT:   0 = none, 1 = silu(bn1), 2 = relu(bn2(silu(bn1)))
// ============================================================================
template<int BMODE, int CMODE, int ACT>
__global__ __launch_bounds__(256) void gemm_wmma(
    const _Float16* __restrict__ A, long astride, int lda,
    const _Float16* __restrict__ B, long bstride, int ldb,
    void* __restrict__ C, long cstride, int ldc,
    int M, int K,
    const float* __restrict__ g1, const float* __restrict__ b1,
    const float* __restrict__ g2, const float* __restrict__ b2)
{
  __shared__ __align__(16) _Float16 As[2][128 * 32];
  __shared__ __align__(16) _Float16 Bs[2][64 * 32];

  const int tid  = threadIdx.x;
  const int bz   = blockIdx.z;
  const int n0   = blockIdx.x * 64;
  const int m0   = blockIdx.y * 128;
  const _Float16* Ab = A + (long)bz * astride;
  const _Float16* Bb = B + (long)bz * bstride;

  const int lane = tid & 31;
  const int wave = tid >> 5;
  const int mo   = (wave >> 1) * 32;   // wave M offset in block tile
  const int no   = (wave & 1) * 32;    // wave N offset in block tile

  // staging-thread coordinates
  const int arow = tid >> 1, aseg = tid & 1;          // A: 2 thr/row, 16 halfs
  const int bkrow = tid >> 3, bcol0 = (tid & 7) * 8;  // B: 8 thr/krow, 8 cols

  auto gloadA = [&](int kk) -> uint4 {
    uint4 v{};
    if (m0 + arow < M)
      v = *(const uint4*)(Ab + (long)(m0 + arow) * lda + kk + aseg * 16);
    return v;
  };
  auto gloadB = [&](int kk, _Float16* o) {
    if (BMODE == 0) {
      uint4 v = *(const uint4*)(Bb + (long)(kk + bkrow) * ldb + n0 + bcol0);
      *(uint4*)o = v;
    } else {
      const int k  = kk + bkrow;
      const int ci = k / 9, r9 = k - ci * 9;
      const int ky = r9 / 3, kx = r9 - ky * 3;
      const int y  = blockIdx.x;                 // N-block == output row
      const int iy = 2 * y + ky - 1;
      const _Float16* src = Bb + ((long)ci * 128 + iy) * 128;
      const bool rowok = (iy >= 0) && (iy < 128);
      #pragma unroll
      for (int j = 0; j < 8; j++) {
        const int ix = 2 * (bcol0 + j) + kx - 1;
        o[j] = (rowok && ix >= 0 && ix < 128) ? src[ix] : (_Float16)0.f;
      }
    }
  };
  auto stageA = [&](int buf, uint4 v) {
    *(uint4*)(&As[buf][arow * 32 + aseg * 16]) = v;
  };
  auto stageB = [&](int buf, const _Float16* o) {
    #pragma unroll
    for (int j = 0; j < 8; j++) Bs[buf][(bcol0 + j) * 32 + bkrow] = o[j];
  };

  v8f acc[2][2] = {};

  // ---- prologue: stage tile 0 into buffer 0 ----
  {
    uint4 a0 = gloadA(0);
    _Float16 b0[8];
    gloadB(0, b0);
    stageA(0, a0);
    stageB(0, b0);
  }
  __syncthreads();

  int cur = 0;
  for (int k0 = 0; k0 < K; k0 += 32) {
    const bool more = (k0 + 32) < K;
    uint4 aN{};
    _Float16 bN[8];
    if (more) {                      // global loads in flight during WMMA
      aN = gloadA(k0 + 32);
      gloadB(k0 + 32, bN);
    }
    if (k0 + 64 < K) {               // prefetch two tiles ahead -> L2
      __builtin_prefetch(Ab + (long)(m0 + arow) * lda + k0 + 64 + aseg * 16, 0, 1);
      if (BMODE == 0)
        __builtin_prefetch(Bb + (long)(k0 + 64 + bkrow) * ldb + n0 + bcol0, 0, 1);
    }

    // ---- build fragments per ISA 7.12.2 wave32 layouts, then 2x2 WMMA ----
    FragH fa[2], fbg[2];
    #pragma unroll
    for (int f = 0; f < 2; f++) {
      const int row = mo + f * 16 + (lane & 15);
      #pragma unroll
      for (int j = 0; j < 8; j++) {
        const int kk = ((j < 4) ? 0 : 16) + ((lane >> 4) << 3) + ((j & 3) << 1);
        fa[f].u[j] = *(const unsigned*)(&As[cur][row * 32 + kk]);
      }
      const int col = no + f * 16 + (lane & 15);
      const int kb  = (lane >> 4) << 3;
      #pragma unroll
      for (int j = 0; j < 8; j++)
        fbg[f].u[j] = *(const unsigned*)(&Bs[cur][col * 32 + ((j < 4) ? 0 : 16) + kb + ((j & 3) << 1)]);
    }
    #pragma unroll
    for (int fm = 0; fm < 2; fm++)
      #pragma unroll
      for (int fn = 0; fn < 2; fn++)
        acc[fm][fn] = __builtin_amdgcn_wmma_f32_16x16x32_f16(
            false, fa[fm].v, false, fbg[fn].v, (short)0, acc[fm][fn], false, false);

    if (more) {                      // stage next tile into the other buffer
      stageA(cur ^ 1, aN);
      stageB(cur ^ 1, bN);
    }
    __syncthreads();
    cur ^= 1;
  }

  // ---- epilogue + store ----
  #pragma unroll
  for (int fm = 0; fm < 2; fm++) {
    #pragma unroll
    for (int fn = 0; fn < 2; fn++) {
      const int col   = n0 + no + fn * 16 + (lane & 15);
      const int mbase = m0 + mo + fm * 16 + ((lane >> 4) << 3);
      #pragma unroll
      for (int r = 0; r < 8; r++) {
        const int m = mbase + r;
        if (m >= M) continue;
        float v = acc[fm][fn][r];
        if (ACT == 1) {
          v = v * g1[m] + b1[m];
          v = v * fast_sigmoid(v);
        } else if (ACT == 2) {
          v = v * g1[m] + b1[m];
          v = v * fast_sigmoid(v);
          v = v * g2[m] + b2[m];
          v = fmaxf(v, 0.f);
        }
        if (CMODE == 0) {
          ((float*)C)[(long)bz * cstride + (long)m * ldc + col] = v;
        } else if (CMODE == 1) {
          ((_Float16*)C)[(long)bz * cstride + (long)m * ldc + col] = (_Float16)v;
        } else {
          // Fourier scatter: bz = tile index (n*288 + ch); m 0-63 low.re,
          // 64-127 high.re, 128-191 low.im, 192-255 high.im
          const int nimg = bz / 288, ch = bz - nimg * 288;
          const int g    = m >> 6;
          const int off  = (g == 0) ? 0 : (g == 1) ? 576 : (g == 2) ? 288 : 864;
          const long a   = (((long)nimg * 1152 + ch + off) * 64 + (m & 63)) * 64 + (col - n0);
          ((_Float16*)C)[a] = (_Float16)v;
        }
      }
    }
  }
}

// ============================================================================
// Elementwise kernels
// ============================================================================
__global__ void k_cvt_f32_f16(const float* __restrict__ s, _Float16* __restrict__ d, long n) {
  long i = (long)blockIdx.x * blockDim.x + threadIdx.x;
  if (i < n) d[i] = (_Float16)s[i];
}

// DFT tables: A2 = [[Re,-Im],[Im,Re]] (256x256), Wre/Wim (128x64) with
// W[c][v] = exp(-2*pi*i*c*v/128)
__global__ void k_tables(_Float16* __restrict__ A2, _Float16* __restrict__ Wre,
                         _Float16* __restrict__ Wim) {
  long i = (long)blockIdx.x * blockDim.x + threadIdx.x;
  const float w = 3.14159265358979323846f / 64.0f;  // 2*pi/128
  if (i < 65536) {
    const int m = (int)(i >> 8), k = (int)(i & 255);
    const int u = m & 127, r = k & 127;
    const float ang = w * (float)((u * r) & 127);
    const float c = __cosf(ang), s = __sinf(ang);
    // Wc_re = cos, Wc_im = -sin :  [[c, +s],[ -s, c]]
    const float val = (m < 128) ? ((k < 128) ? c : s) : ((k < 128) ? -s : c);
    A2[i] = (_Float16)val;
  } else if (i < 73728) {
    const long t = i - 65536; const int c = (int)(t >> 6), v = (int)(t & 63);
    Wre[t] = (_Float16)__cosf(w * (float)((c * v) & 127));
  } else if (i < 81920) {
    const long t = i - 73728; const int c = (int)(t >> 6), v = (int)(t & 63);
    Wim[t] = (_Float16)(-__sinf(w * (float)((c * v) & 127)));
  }
}

// spd: x (8,64,256,256) -> xs_h (8,256,128,128) f16; q order: (0,0),(1,0),(0,1),(1,1)
__global__ void k_spd(const float* __restrict__ x, _Float16* __restrict__ xs, long total) {
  long i = (long)blockIdx.x * blockDim.x + threadIdx.x;
  if (i >= total) return;
  const int j  = (int)(i & 127);
  const int ii = (int)((i >> 7) & 127);
  const int cq = (int)((i >> 14) & 255);
  const int n  = (int)(i >> 22);
  const int c  = cq & 63, q = cq >> 6;
  const int dy = q & 1, dx = q >> 1;
  xs[i] = (_Float16)x[(((long)n * 64 + c) * 256 + 2 * ii + dy) * 256 + 2 * j + dx];
}

// 2x2 avg-pool of w0h channels [0,288) with zero pad to 128x128 -> x1p f16
__global__ void k_pool_x1(const _Float16* __restrict__ w0h, _Float16* __restrict__ x1p, long total) {
  long i = (long)blockIdx.x * blockDim.x + threadIdx.x;
  if (i >= total) return;
  const int j  = (int)(i & 127);
  const int ii = (int)((i >> 7) & 127);
  long rest = i >> 14;
  const int ch = (int)(rest % 288);
  const int n  = (int)(rest / 288);
  _Float16 out = (_Float16)0.f;
  if (ii < 127 && j < 127) {
    const _Float16* p = w0h + (((long)n * 576 + ch) * 128 + ii) * 128 + j;
    out = (_Float16)(0.25f * ((float)p[0] + (float)p[1] + (float)p[128] + (float)p[129]));
  }
  x1p[i] = out;
}

// fused 2x2 avg-pool + 3x3 s2 p1 maxpool on w0h channels [288,576) -> x2p f16
__global__ void k_mpool_x2(const _Float16* __restrict__ w0h, _Float16* __restrict__ x2p, long total) {
  long i = (long)blockIdx.x * blockDim.x + threadIdx.x;
  if (i >= total) return;
  const int xx = (int)(i & 63);
  const int yy = (int)((i >> 6) & 63);
  long rest = i >> 12;
  const int ch = (int)(rest % 288);
  const int n  = (int)(rest / 288);
  const _Float16* base = w0h + (((long)n * 576 + 288 + ch) * 128) * 128;
  float best = -1e30f;
  #pragma unroll
  for (int dr = 0; dr < 3; dr++) {
    const int r = 2 * yy - 1 + dr;
    if (r < 0 || r >= 127) continue;
    #pragma unroll
    for (int dc = 0; dc < 3; dc++) {
      const int c = 2 * xx - 1 + dc;
      if (c < 0 || c >= 127) continue;
      const _Float16* p = base + r * 128 + c;
      const float v = 0.25f * ((float)p[0] + (float)p[1] + (float)p[128] + (float)p[129]);
      best = fmaxf(best, v);
    }
  }
  x2p[i] = (_Float16)best;
}

// wd[c*9+j] = f[c*9+j] * softmax_j(weight[c*9+j])  -> f16
__global__ void k_softmax_wd(const float* __restrict__ wt, const float* __restrict__ f,
                             _Float16* __restrict__ wd, long total) {
  long i = (long)blockIdx.x * blockDim.x + threadIdx.x;
  if (i >= total) return;
  const int px = (int)(i & 4095);
  long rest = i >> 12;
  const int c = (int)(rest & 63);
  const int n = (int)(rest >> 6);
  const long base = (((long)n * 576 + c * 9) * 4096) + px;
  float w[9], m = -1e30f;
  #pragma unroll
  for (int j = 0; j < 9; j++) { w[j] = wt[base + (long)j * 4096]; m = fmaxf(m, w[j]); }
  float s = 0.f;
  #pragma unroll
  for (int j = 0; j < 9; j++) { w[j] = __expf(w[j] - m); s += w[j]; }
  const float inv = __builtin_amdgcn_rcpf(s);
  #pragma unroll
  for (int j = 0; j < 9; j++)
    wd[base + (long)j * 4096] = (_Float16)(f[base + (long)j * 4096] * w[j] * inv);
}

// ============================================================================
extern "C" void kernel_launch(void* const* d_in, const int* in_sizes, int n_in,
                              void* d_out, int out_size, void* d_ws, size_t ws_size,
                              hipStream_t stream) {
  const float* x     = (const float*)d_in[0];
  const float* w_gw  = (const float*)d_in[1];
  const float* g_gw  = (const float*)d_in[2];
  const float* b_gw  = (const float*)d_in[3];
  const float* w_fm  = (const float*)d_in[4];
  const float* g_fm  = (const float*)d_in[5];
  const float* b_fm  = (const float*)d_in[6];
  const float* w_cv2 = (const float*)d_in[7];
  const float* g_cv2 = (const float*)d_in[8];
  const float* b_cv2 = (const float*)d_in[9];
  const float* w_gf  = (const float*)d_in[10];
  const float* g_gf  = (const float*)d_in[11];
  const float* b_gf  = (const float*)d_in[12];
  const float* g_gf2 = (const float*)d_in[13];
  const float* b_gf2 = (const float*)d_in[14];
  const float* w_out = (const float*)d_in[15];
  float* out = (float*)d_out;

  char* p = (char*)d_ws;
  auto carve = [&](size_t bytes) { char* r = p; p += (bytes + 255) & ~(size_t)255; return r; };
  _Float16* xs_h  = (_Float16*)carve(8L * 256 * 128 * 128 * 2);
  _Float16* w0h   = (_Float16*)carve(8L * 576 * 128 * 128 * 2);
  _Float16* x1p   = (_Float16*)carve(8L * 288 * 128 * 128 * 2);
  _Float16* Tbuf  = (_Float16*)carve(2304L * 256 * 64 * 2);
  _Float16* cat   = (_Float16*)carve(8L * 1152 * 64 * 64 * 2);
  _Float16* x2p   = (_Float16*)carve(8L * 288 * 64 * 64 * 2);
  float*    wtbuf = (float*)   carve(8L * 576 * 4096 * 4);
  float*    fbuf  = (float*)   carve(8L * 576 * 4096 * 4);
  _Float16* wd    = (_Float16*)carve(8L * 576 * 4096 * 2);
  _Float16* wgwh  = (_Float16*)carve(576L * 256 * 2);
  _Float16* wfmh  = (_Float16*)carve(288L * 1152 * 2);
  _Float16* wcvh  = (_Float16*)carve(288L * 288 * 2);
  _Float16* wgfh  = (_Float16*)carve(576L * 2304 * 2);
  _Float16* wouth = (_Float16*)carve(128L * 576 * 2);
  _Float16* A2    = (_Float16*)carve(256L * 256 * 2);
  _Float16* Wre   = (_Float16*)carve(128L * 64 * 2);
  _Float16* Wim   = (_Float16*)carve(128L * 64 * 2);

  auto nb = [](long n) { return (unsigned)((n + 255) / 256); };

  // ---- init: DFT tables + f16 weight copies ----
  k_tables<<<320, 256, 0, stream>>>(A2, Wre, Wim);
  k_cvt_f32_f16<<<nb(576L * 256), 256, 0, stream>>>(w_gw, wgwh, 576L * 256);
  k_cvt_f32_f16<<<nb(288L * 1152), 256, 0, stream>>>(w_fm, wfmh, 288L * 1152);
  k_cvt_f32_f16<<<nb(288L * 288), 256, 0, stream>>>(w_cv2, wcvh, 288L * 288);
  k_cvt_f32_f16<<<nb(576L * 2304), 256, 0, stream>>>(w_gf, wgfh, 576L * 2304);
  k_cvt_f32_f16<<<nb(128L * 576), 256, 0, stream>>>(w_out, wouth, 128L * 576);

  // ---- space-to-depth pack (also the f16 operand for both big convs) ----
  k_spd<<<nb(8L * 256 * 128 * 128), 256, 0, stream>>>(x, xs_h, 8L * 256 * 128 * 128);

  // ---- G1: w0 = silu(bn(W_gw(576x256) @ xs)), per batch N=16384 ----
  gemm_wmma<0, 1, 1><<<dim3(256, 5, 8), 256, 0, stream>>>(
      wgwh, 0, 256, xs_h, 256L * 16384, 16384,
      (void*)w0h, 576L * 16384, 16384, 576, 256, g_gw, b_gw, nullptr, nullptr);

  // ---- pooling branches ----
  k_pool_x1 <<<nb(8L * 288 * 128 * 128), 256, 0, stream>>>(w0h, x1p, 8L * 288 * 128 * 128);
  k_mpool_x2<<<nb(8L * 288 * 64 * 64), 256, 0, stream>>>(w0h, x2p, 8L * 288 * 64 * 64);

  // ---- rfft2 as DFT GEMMs over 2304 tiles ----
  // row transform: T_re rows [0,128), T_im rows [128,256) of Tbuf (256x64/tile)
  gemm_wmma<0, 1, 0><<<dim3(1, 1, 2304), 256, 0, stream>>>(
      x1p, 16384, 128, Wre, 0, 64, (void*)Tbuf, 16384, 64, 128, 128,
      nullptr, nullptr, nullptr, nullptr);
  gemm_wmma<0, 1, 0><<<dim3(1, 1, 2304), 256, 0, stream>>>(
      x1p, 16384, 128, Wim, 0, 64, (void*)(Tbuf + 8192), 16384, 64, 128, 128,
      nullptr, nullptr, nullptr, nullptr);
  // column transform + scatter into cat (8,1152,64,64)
  gemm_wmma<0, 2, 0><<<dim3(1, 2, 2304), 256, 0, stream>>>(
      A2, 0, 256, Tbuf, 16384, 64, (void*)cat, 0, 0, 256, 256,
      nullptr, nullptr, nullptr, nullptr);

  // ---- x1 = silu(bn(W_fm @ cat)) -> weight[:,0:288] ----
  gemm_wmma<0, 0, 1><<<dim3(64, 3, 8), 256, 0, stream>>>(
      wfmh, 0, 1152, cat, 1152L * 4096, 4096,
      (void*)wtbuf, 576L * 4096, 4096, 288, 1152, g_fm, b_fm, nullptr, nullptr);
  // ---- x2 = silu(bn(W_cv2 @ x2p)) -> weight[:,288:576] ----
  gemm_wmma<0, 0, 1><<<dim3(64, 3, 8), 256, 0, stream>>>(
      wcvh, 0, 288, x2p, 288L * 4096, 4096,
      (void*)(wtbuf + 288L * 4096), 576L * 4096, 4096, 288, 288,
      g_cv2, b_cv2, nullptr, nullptr);

  // ---- f = relu(bn2(silu(bn1(conv3x3s2(xs))))) — implicit GEMM, K=2304 ----
  gemm_wmma<1, 0, 2><<<dim3(64, 5, 8), 256, 0, stream>>>(
      wgfh, 0, 2304, xs_h, 256L * 16384, 0,
      (void*)fbuf, 576L * 4096, 4096, 576, 2304, g_gf, b_gf, g_gf2, b_gf2);

  // ---- wd = f * softmax_9(weight) ----
  k_softmax_wd<<<nb(8L * 64 * 4096), 256, 0, stream>>>(wtbuf, fbuf, wd, 8L * 64 * 4096);

  // ---- out = W_out(128x576) @ wd  (pixel-shuffle + s3 conv collapsed) ----
  gemm_wmma<0, 0, 0><<<dim3(64, 1, 8), 256, 0, stream>>>(
      wouth, 0, 576, wd, 576L * 4096, 4096,
      (void*)out, 128L * 4096, 4096, 128, 576, nullptr, nullptr, nullptr, nullptr);
}
